// TransformerBlock_65584150609955
// MI455X (gfx1250) — matrix operand order repaired and checked by reading, not compiled, not run on hardware
//
#include <hip/hip_runtime.h>
#include <hip/hip_bf16.h>
#include <math.h>

// ---------------------------------------------------------------------------
// Transformer block forward for MI455X (gfx1250), bf16 WMMA + f32 accumulate.
// B=8, L=1024, D=1024, H=16, DH=64, Dff=4096, M=B*L=8192.
// Uses v_wmma_f32_16x16x32_bf16 for all GEMMs/attention and
// global_load_async_to_lds_b128 (ASYNCcnt) for the copy-only LDS staging.
// ---------------------------------------------------------------------------

typedef __bf16 bf16;
typedef __attribute__((ext_vector_type(16))) __bf16 v16bf;
typedef __attribute__((ext_vector_type(8)))  float  v8f;

#define WMMA_BF16(a, b, c) \
  __builtin_amdgcn_wmma_f32_16x16x32_bf16(false, (a), false, (b), (short)0, (c), false, false)

// async global->LDS copy of 16B per lane; completion via ASYNCcnt
#define ASYNC_LOAD_B128(ldsaddr32, gaddr64)                      \
  asm volatile("global_load_async_to_lds_b128 %0, %1, off"       \
               :: "v"(ldsaddr32), "v"(gaddr64) : "memory")
#define WAIT_ASYNC0() asm volatile("s_wait_asynccnt 0x0" ::: "memory")

#define LSTR 40   // LDS row stride (80B) for 32-element rows
#define KSTR 72   // LDS row stride (144B) for 64-element rows

union Frag16 { v16bf v; uint4 u[2]; };

// ---------------------------------------------------------------------------
// fp32 -> bf16 elementwise conversion (weights, once per launch)
// ---------------------------------------------------------------------------
__global__ __launch_bounds__(256) void cvt_f32_bf16_kernel(
    const float* __restrict__ in, bf16* __restrict__ out, int n) {
  int i = blockIdx.x * blockDim.x + threadIdx.x;
  int stride = gridDim.x * blockDim.x;
  for (; i < n; i += stride) out[i] = (bf16)in[i];
}

// ---------------------------------------------------------------------------
// LayerNorm over D=1024, one block (256 threads) per row, bf16 output.
// ---------------------------------------------------------------------------
__global__ __launch_bounds__(256) void layernorm_kernel(
    const float* __restrict__ x, const float* __restrict__ w,
    const float* __restrict__ b, bf16* __restrict__ out) {
  __shared__ float red[16];
  const int row = blockIdx.x;
  const int t = threadIdx.x;
  const float* xr = x + (size_t)row * 1024;
  float4 v = *(const float4*)(xr + t * 4);
  float s1 = v.x + v.y + v.z + v.w;
  float s2 = v.x * v.x + v.y * v.y + v.z * v.z + v.w * v.w;
#pragma unroll
  for (int m = 16; m >= 1; m >>= 1) {
    s1 += __shfl_xor(s1, m, 32);
    s2 += __shfl_xor(s2, m, 32);
  }
  const int wave = t >> 5, lane = t & 31;
  if (lane == 0) { red[wave] = s1; red[8 + wave] = s2; }
  __syncthreads();
  if (t == 0) {
    float a = 0.f, c = 0.f;
#pragma unroll
    for (int i = 0; i < 8; ++i) { a += red[i]; c += red[8 + i]; }
    red[0] = a; red[8] = c;
  }
  __syncthreads();
  const float mean = red[0] * (1.0f / 1024.0f);
  const float var  = red[8] * (1.0f / 1024.0f) - mean * mean;
  const float inv  = rsqrtf(var + 1e-12f);
  float4 wv = *(const float4*)(w + t * 4);
  float4 bv = *(const float4*)(b + t * 4);
  bf16* o = out + (size_t)row * 1024 + t * 4;
  o[0] = (bf16)(wv.x * ((v.x - mean) * inv) + bv.x);
  o[1] = (bf16)(wv.y * ((v.y - mean) * inv) + bv.y);
  o[2] = (bf16)(wv.z * ((v.z - mean) * inv) + bv.z);
  o[3] = (bf16)(wv.w * ((v.w - mean) * inv) + bv.w);
}

// ---------------------------------------------------------------------------
// Tiled bf16 WMMA GEMM:  C[M,N] = oscale*(A[M,K]*W[K,N] + bias) (+res, GELU)
// Block = 256 threads (8 waves) computes a 256x64 tile; K steps of 32.
// A tile staged via async global->LDS DMA; B tile transposed through VGPRs.
// Wave w owns rows 32w..32w+31: 8 WMMAs per K-step, B fragment reused twice.
// ---------------------------------------------------------------------------
template <bool GELU, bool SCATTER_BHLD, bool HAS_RES, bool OUTF>
__global__ __launch_bounds__(256) void gemm_bf16_kernel(
    const bf16* __restrict__ A, const bf16* __restrict__ W,
    const float* __restrict__ bias, const float* __restrict__ residual,
    float* __restrict__ outF, bf16* __restrict__ outB,
    float oscale, int N, int K) {
  __shared__ bf16 As[256 * LSTR];   // A tile, row-major [256][32]
  __shared__ bf16 Bs[64 * LSTR];    // B tile transposed [64 cols][32 k]

  const int t = threadIdx.x;
  const int lane = t & 31;
  const int wave = t >> 5;
  const int rowBase = blockIdx.x * 256;
  const int colBase = blockIdx.y * 64;
  const unsigned asBase = (unsigned)(uintptr_t)(&As[0]);

  const v8f zero = {0.f, 0.f, 0.f, 0.f, 0.f, 0.f, 0.f, 0.f};
  v8f acc[2][4] = {{zero, zero, zero, zero}, {zero, zero, zero, zero}};

  for (int kk = 0; kk < K; kk += 32) {
    __syncthreads();
    // -- A tile fill via async DMA: 256x32 bf16 = 1024 x b128, 4 per thread
#pragma unroll
    for (int i = 0; i < 4; ++i) {
      const int c = t + i * 256;
      const int r = c >> 2;
      const int cc = (c & 3) * 8;
      const unsigned long long ga =
          (unsigned long long)(uintptr_t)(A + (size_t)(rowBase + r) * K + kk + cc);
      const unsigned la = asBase + (unsigned)((r * LSTR + cc) * 2);
      ASYNC_LOAD_B128(la, ga);
    }
    // -- B tile fill (transposed through VGPRs): 32x64 bf16 = 256 uint4
    {
      const int kr = t >> 3;
      const int cc = (t & 7) * 8;
      uint4 val = *(const uint4*)(W + (size_t)(kk + kr) * N + colBase + cc);
      bf16 tmp[8];
      *(uint4*)tmp = val;
#pragma unroll
      for (int j = 0; j < 8; ++j) Bs[(cc + j) * LSTR + kr] = tmp[j];
    }
    if (kk + 32 < K) {  // gfx1250 global_prefetch_b8 for the next tiles
      __builtin_prefetch(A + (size_t)(rowBase + t) * K + kk + 32, 0, 0);
      __builtin_prefetch(W + (size_t)(kk + 32 + (t >> 3)) * N + colBase + (t & 7) * 8, 0, 0);
    }
    WAIT_ASYNC0();       // async A-tile writes landed in LDS
    __syncthreads();

    // -- two A fragments (rows 32w+0..15, 32w+16..31)
    Frag16 af[2];
#pragma unroll
    for (int h = 0; h < 2; ++h) {
      const int ar = wave * 32 + h * 16 + (lane & 15);
      const int ac = (lane >> 4) * 8;
      af[h].u[0] = *(const uint4*)(&As[ar * LSTR + ac]);
      af[h].u[1] = *(const uint4*)(&As[ar * LSTR + ac + 16]);
    }
#pragma unroll
    for (int nt = 0; nt < 4; ++nt) {
      Frag16 bfr;  // B fragment: lane = column, lane half selects K 8h..8h+7 and 16+8h..16+8h+7
      const int bc = nt * 16 + (lane & 15);
      const int bk = (lane >> 4) * 8;
      bfr.u[0] = *(const uint4*)(&Bs[bc * LSTR + bk]);
      bfr.u[1] = *(const uint4*)(&Bs[bc * LSTR + bk + 16]);
#pragma unroll
      for (int h = 0; h < 2; ++h)
        acc[h][nt] = WMMA_BF16(af[h].v, bfr.v, acc[h][nt]);
    }
  }

  // ---- epilogue: C layout is N = lane%16, M = r + 8*(lane/16)
#pragma unroll
  for (int h = 0; h < 2; ++h) {
#pragma unroll
    for (int nt = 0; nt < 4; ++nt) {
#pragma unroll
      for (int r = 0; r < 8; ++r) {
        const int row = rowBase + wave * 32 + h * 16 + r + 8 * (lane >> 4);
        const int col = colBase + nt * 16 + (lane & 15);
        float v = (acc[h][nt][r] + bias[col]) * oscale;
        if (GELU) {
          const float g = v;
          v = 0.5f * g * (1.0f + tanhf(0.7978845608028654f * (g + 0.044715f * g * g * g)));
        }
        if (HAS_RES) v += residual[(size_t)row * N + col];
        if (OUTF) {
          outF[(size_t)row * N + col] = v;
        } else if (SCATTER_BHLD) {  // (B,L,D) output -> (B,H,L,DH)
          const int b = row >> 10, q = row & 1023;
          const int hh = col >> 6, d = col & 63;
          outB[((((size_t)b * 16 + hh) << 10) + q) * 64 + d] = (bf16)v;
        } else {
          outB[(size_t)row * N + col] = (bf16)v;
        }
      }
    }
  }
}

// ---------------------------------------------------------------------------
// Flash-attention: grid (B*H, L/128). 8 waves/block, one 16-query tile per
// wave. All waves iterate the same uniform 32-chunk loop; K chunk staged via
// async global->LDS DMA, V chunk transposed through VGPRs. Q is pre-scaled
// by 1/sqrt(DH) at projection time. Branchless masks handle both reverse
// directions and reproduce the pad-masked uniform softmax.
// ---------------------------------------------------------------------------
__global__ __launch_bounds__(256) void attention_kernel(
    const bf16* __restrict__ Q, const bf16* __restrict__ Kt,
    const bf16* __restrict__ V, const unsigned char* __restrict__ mask,
    const int* __restrict__ reversep, bf16* __restrict__ ctx) {
  __shared__ bf16 Ks[32 * KSTR];        // K chunk  [32 keys][64 dh]
  __shared__ bf16 Vt[64 * LSTR];        // V chunk transposed [64 dh][32 keys]
  __shared__ bf16 Pst[8][16 * LSTR];    // per-wave P staging [16 q][32 keys]

  const int bh = blockIdx.x;     // 0..127 = b*16 + h
  const int bidx = bh >> 4;
  const int hh = bh & 15;
  const int t = threadIdx.x, lane = t & 31, wave = t >> 5;
  const int qbase = blockIdx.y * 128 + wave * 16;
  const int sgn = 1 - 2 * reversep[0];   // +1: mask q<k ; -1: mask q>k
  const unsigned ksBase = (unsigned)(uintptr_t)(&Ks[0]);

  const bf16* Qb = Q + (size_t)bh * 1024 * 64;
  const bf16* Kb = Kt + (size_t)bh * 1024 * 64;
  const bf16* Vb = V + (size_t)bh * 1024 * 64;

  // persistent Q A-fragments for dh 0-31 and 32-63 (Q already has 1/8 folded)
  Frag16 qf[2];
  const int qrow = qbase + (lane & 15);
#pragma unroll
  for (int s = 0; s < 2; ++s) {
    const int c1 = s * 32 + (lane >> 4) * 8;
    qf[s].u[0] = *(const uint4*)(Qb + (size_t)qrow * 64 + c1);
    qf[s].u[1] = *(const uint4*)(Qb + (size_t)qrow * 64 + c1 + 16);
  }

  const v8f zero = {0.f, 0.f, 0.f, 0.f, 0.f, 0.f, 0.f, 0.f};
  v8f acc[4] = {zero, zero, zero, zero};
  float m_run[8], l_run[8];
  int myq[8];
  bool pad[8];
#pragma unroll
  for (int r = 0; r < 8; ++r) {
    m_run[r] = -3.0e38f;
    l_run[r] = 0.f;
    myq[r] = qbase + r + 8 * (lane >> 4);
    pad[r] = mask[(size_t)bidx * 1024 + myq[r]] != 0;
  }

  for (int c = 0; c < 32; ++c) {  // 32 chunks x 32 keys = L
    const int kb = c * 32;
    __syncthreads();
    // ---- stage K chunk row-major via async DMA: 256 x b128, 1 per thread
    {
      const int kr = t >> 3;
      const int cc = (t & 7) * 8;
      const unsigned long long ga =
          (unsigned long long)(uintptr_t)(Kb + (size_t)(kb + kr) * 64 + cc);
      const unsigned la = ksBase + (unsigned)((kr * KSTR + cc) * 2);
      ASYNC_LOAD_B128(la, ga);
    }
    // ---- stage V chunk transposed: Vt[dh][key]
    {
      const int key = t >> 3;
      const int d0 = (t & 7) * 8;
      uint4 val = *(const uint4*)(Vb + (size_t)(kb + key) * 64 + d0);
      bf16 tmp[8];
      *(uint4*)tmp = val;
#pragma unroll
      for (int j = 0; j < 8; ++j) Vt[(d0 + j) * LSTR + key] = tmp[j];
    }
    WAIT_ASYNC0();
    __syncthreads();

    // ---- S = (Q/sqrt(DH)) K^T: 2 key sub-tiles x 2 dh halves = 4 WMMA
    v8f s0 = zero, s1 = zero;
#pragma unroll
    for (int tkv = 0; tkv < 2; ++tkv) {
      const int krow = tkv * 16 + (lane & 15);
#pragma unroll
      for (int s = 0; s < 2; ++s) {
        Frag16 kf;  // B frag: column = key, contiguous dh in LDS
        const int d0 = s * 32 + (lane >> 4) * 8;
        kf.u[0] = *(const uint4*)(&Ks[krow * KSTR + d0]);
        kf.u[1] = *(const uint4*)(&Ks[krow * KSTR + d0 + 16]);
        if (tkv == 0) s0 = WMMA_BF16(qf[s].v, kf.v, s0);
        else          s1 = WMMA_BF16(qf[s].v, kf.v, s1);
      }
    }
    // ---- branchless masks
#pragma unroll
    for (int r = 0; r < 8; ++r) {
      const int q = myq[r];
      const int k0 = kb + (lane & 15);
      const int k1 = k0 + 16;
      const bool m0 = pad[r] | (sgn * (k0 - q) > 0);
      const bool m1 = pad[r] | (sgn * (k1 - q) > 0);
      s0[r] = m0 ? -1.0e9f : s0[r];
      s1[r] = m1 ? -1.0e9f : s1[r];
    }
    // ---- online softmax (row = 16 lanes of one half)
    float mnew[8], alpha[8];
#pragma unroll
    for (int r = 0; r < 8; ++r) {
      float mx = fmaxf(s0[r], s1[r]);
      mx = fmaxf(mx, __shfl_xor(mx, 1, 32));
      mx = fmaxf(mx, __shfl_xor(mx, 2, 32));
      mx = fmaxf(mx, __shfl_xor(mx, 4, 32));
      mx = fmaxf(mx, __shfl_xor(mx, 8, 32));
      mnew[r] = fmaxf(m_run[r], mx);
      alpha[r] = __expf(m_run[r] - mnew[r]);
      m_run[r] = mnew[r];
    }
#pragma unroll
    for (int r = 0; r < 8; ++r) {
      const float p0 = __expf(s0[r] - mnew[r]);
      const float p1 = __expf(s1[r] - mnew[r]);
      s0[r] = p0; s1[r] = p1;
      float rs = p0 + p1;
      rs += __shfl_xor(rs, 1, 32);
      rs += __shfl_xor(rs, 2, 32);
      rs += __shfl_xor(rs, 4, 32);
      rs += __shfl_xor(rs, 8, 32);
      l_run[r] = l_run[r] * alpha[r] + rs;
    }
#pragma unroll
    for (int nt = 0; nt < 4; ++nt)
#pragma unroll
      for (int r = 0; r < 8; ++r) acc[nt][r] *= alpha[r];

    // ---- stage P (bf16) to per-wave LDS in (row, key) order
    bf16* P = &Pst[wave][0];
    const int prow = 8 * (lane >> 4);
    const int pcol = lane & 15;
#pragma unroll
    for (int r = 0; r < 8; ++r) {
      P[(prow + r) * LSTR + pcol]      = (bf16)s0[r];
      P[(prow + r) * LSTR + 16 + pcol] = (bf16)s1[r];
    }
    // same-wave DS ops complete in order; explicit wait keeps ordering tight
    asm volatile("s_wait_dscnt 0" ::: "memory");

    // ---- P A-fragment (16x32 over keys)
    Frag16 pf;
    const int pr = lane & 15;
    const int pc1 = (lane >> 4) * 8;
    pf.u[0] = *(const uint4*)(&P[pr * LSTR + pc1]);
    pf.u[1] = *(const uint4*)(&P[pr * LSTR + pc1 + 16]);

    // ---- O += P @ V : 4 dh sub-tiles, V fragments from transposed LDS
#pragma unroll
    for (int nt = 0; nt < 4; ++nt) {
      Frag16 vf;
      const int d = nt * 16 + (lane & 15);
      const int k0 = (lane >> 4) * 8;
      vf.u[0] = *(const uint4*)(&Vt[d * LSTR + k0]);
      vf.u[1] = *(const uint4*)(&Vt[d * LSTR + k0 + 16]);
      acc[nt] = WMMA_BF16(pf.v, vf.v, acc[nt]);
    }
  }

  // ---- normalize, write ctx as (B, L, H, DH) == (B, L, D) for Wo GEMM
#pragma unroll
  for (int nt = 0; nt < 4; ++nt) {
#pragma unroll
    for (int r = 0; r < 8; ++r) {
      const float o = acc[nt][r] / l_run[r];
      const int q = myq[r];
      const int d = nt * 16 + (lane & 15);
      ctx[(((size_t)bidx * 1024 + q) * 16 + hh) * 64 + d] = (bf16)o;
    }
  }
}

// ---------------------------------------------------------------------------
// Host-side orchestration
// ---------------------------------------------------------------------------
extern "C" void kernel_launch(void* const* d_in, const int* in_sizes, int n_in,
                              void* d_out, int out_size, void* d_ws, size_t ws_size,
                              hipStream_t stream) {
  (void)in_sizes; (void)n_in; (void)out_size;
  const float* hidden = (const float*)d_in[0];
  const unsigned char* mask = (const unsigned char*)d_in[1];
  const int* reverse = (const int*)d_in[2];
  const float* Wq = (const float*)d_in[3];  const float* bq = (const float*)d_in[4];
  const float* Wk = (const float*)d_in[5];  const float* bk = (const float*)d_in[6];
  const float* Wv = (const float*)d_in[7];  const float* bv = (const float*)d_in[8];
  const float* Wo = (const float*)d_in[9];  const float* bo = (const float*)d_in[10];
  const float* W1 = (const float*)d_in[11]; const float* b1 = (const float*)d_in[12];
  const float* W2 = (const float*)d_in[13]; const float* b2 = (const float*)d_in[14];
  const float* ln1w = (const float*)d_in[15]; const float* ln1b = (const float*)d_in[16];
  const float* ln2w = (const float*)d_in[17]; const float* ln2b = (const float*)d_in[18];

  const int Mrows = 8192, D = 1024, Dff = 4096;
  const size_t MB = 1024u * 1024u;

  char* w = (char*)d_ws;
  bf16* h1  = (bf16*)(w + 0);          // [8192,1024] bf16   16 MB
  bf16* qB  = (bf16*)(w + 16 * MB);    // (B,H,L,DH)         16 MB
  bf16* kB  = (bf16*)(w + 32 * MB);    //                    16 MB
  bf16* vB  = (bf16*)(w + 48 * MB);    //                    16 MB
  bf16* ctx = (bf16*)(w + 64 * MB);    // [8192,1024] bf16   16 MB
  float* x2 = (float*)(w + 80 * MB);   // [8192,1024] f32    32 MB
  bf16* h2  = (bf16*)(w + 112 * MB);   // [8192,1024] bf16   16 MB
  bf16* f1  = (bf16*)(w + 128 * MB);   // [8192,4096] bf16   64 MB
  bf16* wqb = (bf16*)(w + 192 * MB);   // 2 MB each
  bf16* wkb = (bf16*)(w + 194 * MB);
  bf16* wvb = (bf16*)(w + 196 * MB);
  bf16* wob = (bf16*)(w + 198 * MB);
  bf16* w1b = (bf16*)(w + 200 * MB);   // 8 MB
  bf16* w2b = (bf16*)(w + 208 * MB);   // 8 MB -> total 216 MB
  if (ws_size < 216 * MB) return;

  // 1) weight conversions fp32 -> bf16
  cvt_f32_bf16_kernel<<<1024, 256, 0, stream>>>(Wq, wqb, D * D);
  cvt_f32_bf16_kernel<<<1024, 256, 0, stream>>>(Wk, wkb, D * D);
  cvt_f32_bf16_kernel<<<1024, 256, 0, stream>>>(Wv, wvb, D * D);
  cvt_f32_bf16_kernel<<<1024, 256, 0, stream>>>(Wo, wob, D * D);
  cvt_f32_bf16_kernel<<<2048, 256, 0, stream>>>(W1, w1b, D * Dff);
  cvt_f32_bf16_kernel<<<2048, 256, 0, stream>>>(W2, w2b, Dff * D);

  // 2) LN1
  layernorm_kernel<<<Mrows, 256, 0, stream>>>(hidden, ln1w, ln1b, h1);

  // 3) Q/K/V projections, scattered to (B,H,L,DH); Q pre-scaled by 1/sqrt(DH)
  dim3 gD(Mrows / 256, D / 64);
  gemm_bf16_kernel<false, true, false, false>
      <<<gD, 256, 0, stream>>>(h1, wqb, bq, nullptr, nullptr, qB, 0.125f, D, D);
  gemm_bf16_kernel<false, true, false, false>
      <<<gD, 256, 0, stream>>>(h1, wkb, bk, nullptr, nullptr, kB, 1.0f, D, D);
  gemm_bf16_kernel<false, true, false, false>
      <<<gD, 256, 0, stream>>>(h1, wvb, bv, nullptr, nullptr, vB, 1.0f, D, D);

  // 4) attention
  attention_kernel<<<dim3(128, 8), 256, 0, stream>>>(qB, kB, vB, mask, reverse, ctx);

  // 5) output projection + residual -> x2 (f32)
  gemm_bf16_kernel<false, false, true, true>
      <<<gD, 256, 0, stream>>>(ctx, wob, bo, hidden, x2, nullptr, 1.0f, D, D);

  // 6) LN2
  layernorm_kernel<<<Mrows, 256, 0, stream>>>(x2, ln2w, ln2b, h2);

  // 7) FFN up + GELU (bf16)
  dim3 gF1(Mrows / 256, Dff / 64);
  gemm_bf16_kernel<true, false, false, false>
      <<<gF1, 256, 0, stream>>>(h2, w1b, b1, nullptr, nullptr, f1, 1.0f, Dff, D);

  // 8) FFN down + residual -> d_out (f32)
  gemm_bf16_kernel<false, false, true, true>
      <<<gD, 256, 0, stream>>>(f1, w2b, b2, x2, (float*)d_out, nullptr, 1.0f, D, Dff);
}
